// GPTQLinear_12360915878783
// MI455X (gfx1250) — matrix operand order repaired and checked by reading, not compiled, not run on hardware
//
#include <hip/hip_runtime.h>
#include <hip/hip_bf16.h>
#include <stdint.h>

// Problem constants (match the reference)
#define M_TOT   256        // B*S = 8*32
#define IN_F    8192
#define OUT_F   8192
#define GROUPSZ 128
#define N_GROUPS (IN_F / GROUPSZ)   // 64

typedef __attribute__((ext_vector_type(16))) _Float16 v16h;
typedef __attribute__((ext_vector_type(8)))  _Float16 v8h;
typedef __attribute__((ext_vector_type(2)))  _Float16 h2;
typedef __attribute__((ext_vector_type(8)))  float    v8f;
typedef __attribute__((ext_vector_type(8)))  uint32_t v8u;

// ---------------------------------------------------------------------------
// Kernel 1: desc_act activation permutation gather.
//   xp[m][k] = x[m][perm[k]]   (fp16, [256][8192] into workspace)
// ---------------------------------------------------------------------------
__global__ __launch_bounds__(256) void gptq_permute_kernel(
    const _Float16* __restrict__ x,
    const long long* __restrict__ perm,
    _Float16* __restrict__ xp)
{
    const int idx = blockIdx.x * blockDim.x + threadIdx.x;   // [0, 256*8192)
    const int m   = idx >> 13;          // / IN_F
    const int k   = idx & (IN_F - 1);
    const int src = (int)perm[k];
    xp[idx] = x[(m << 13) + src];
}

// ---------------------------------------------------------------------------
// int4 -> fp16 dequant, two nibbles at a time, no f32 round-trip.
//   t[3:0]=nib_lo, t[7:4]=nib_hi.  x=(t<<12)|t puts nib_lo at [3:0] and
//   nib_hi at [19:16]; mask + OR 0x6400 magic makes halfs = 1024+nib (exact).
//   (u - 1032) * s  ==  (nib - 8) * s   via v_pk_add_f16 + v_pk_mul_f16.
// ---------------------------------------------------------------------------
__device__ __forceinline__ uint32_t dq_pair(uint32_t t, h2 s2) {
    const uint32_t u = (((t << 12) | t) & 0x000F000Fu) | 0x64006400u;
    h2 v = __builtin_bit_cast(h2, u);
    h2 off = {(_Float16)1032.0f, (_Float16)1032.0f};
    h2 r = (v - off) * s2;
    return __builtin_bit_cast(uint32_t, r);
}

__device__ __forceinline__ v16h dequant16(uint32_t w0, uint32_t w1, h2 s2) {
    v8u r;
    r[0] = dq_pair(w0,        s2);
    r[1] = dq_pair(w0 >> 8,   s2);
    r[2] = dq_pair(w0 >> 16,  s2);
    r[3] = dq_pair(w0 >> 24,  s2);
    r[4] = dq_pair(w1,        s2);
    r[5] = dq_pair(w1 >> 8,   s2);
    r[6] = dq_pair(w1 >> 16,  s2);
    r[7] = dq_pair(w1 >> 24,  s2);
    return __builtin_bit_cast(v16h, r);
}

// ---------------------------------------------------------------------------
// Kernel 2: fused int4-dequant + fp16 WMMA GEMM.
// Workgroup: 256 threads = 8 waves (wave32), 2x4 (M x N) wave grid.
//   WG tile   : 128(M) x 128(N)
//   Wave tile :  64(M) x  32(N) = 4 x 2 WMMA 16x16 tiles
// ---------------------------------------------------------------------------
__global__ __launch_bounds__(256) void gptq_wmma_gemm_kernel(
    const _Float16* __restrict__ xp,   // [M_TOT][IN_F]  permuted activations
    const int*      __restrict__ qw,   // [IN_F/8][OUT_F] packed int4
    const _Float16* __restrict__ sc,   // [N_GROUPS][OUT_F]
    float*          __restrict__ out)  // [M_TOT][OUT_F]
{
    const int lane = threadIdx.x & 31;
    const int wave = threadIdx.x >> 5;      // 0..7
    const int half = lane >> 4;             // 0/1
    const int ln   = lane & 15;

    const int wm = wave >> 2;               // 0..1  (M direction)
    const int wn = wave & 3;                // 0..3  (N direction)

    const int m0 = blockIdx.y * 128 + wm * 64;   // wave M base (4 tiles of 16)
    const int n0 = blockIdx.x * 128 + wn * 32;   // wave N base (2 tiles of 16)

    v8f acc[4][2] = {};

    // A row base pointers per m-tile. 16-bit A 16x32 layout (matches B):
    //   element j of v16h holds K = half*16 + j, row = ln
    // -> two contiguous 16B runs at k0+half*16 and k0+half*16+8.
    const _Float16* ap[4];
#pragma unroll
    for (int t = 0; t < 4; ++t)
        ap[t] = xp + (size_t)(m0 + t * 16 + ln) * IN_F + half * 16;

    // B column pointers. 16-bit B 32x16 layout used here (matches A):
    //   element j holds K = half*16 + j, column n = ln
    // -> exactly two packed int32 qweight words per lane per K-step.
    const int n_col0 = n0 + ln;
    const int n_col1 = n0 + 16 + ln;
    const int* qp0 = qw + (size_t)(half * 2) * OUT_F + n_col0;
    const int* qp1 = qw + (size_t)(half * 2) * OUT_F + n_col1;
    const _Float16* sp0 = sc + n_col0;
    const _Float16* sp1 = sc + n_col1;

    for (int g = 0; g < N_GROUPS; ++g) {
        const _Float16 s0 = sp0[(size_t)g * OUT_F];
        const _Float16 s1 = sp1[(size_t)g * OUT_F];
        const h2 s20 = {s0, s0};
        const h2 s21 = {s1, s1};

        // prefetch next group's qweight block (dominant streamed operand)
        if (g + 1 < N_GROUPS) {
            __builtin_prefetch(qp0 + (size_t)((g + 1) * 16) * OUT_F, 0, 0);
            __builtin_prefetch(qp1 + (size_t)((g + 1) * 16) * OUT_F, 0, 0);
        }

#pragma unroll
        for (int kk = 0; kk < 4; ++kk) {
            const int koff = g * 128 + kk * 32;       // K element offset

            // ---- A fragments: 2x global_load_b128 per m-tile ----
            v16h a[4];
#pragma unroll
            for (int t = 0; t < 4; ++t) {
                v8h lo = *(const v8h*)(ap[t] + koff);
                v8h hi = *(const v8h*)(ap[t] + koff + 8);
                a[t] = __builtin_shufflevector(lo, hi,
                        0, 1, 2, 3, 4, 5, 6, 7, 8, 9, 10, 11, 12, 13, 14, 15);
            }

            // ---- B fragments: on-the-fly int4 -> fp16 dequant ----
            const size_t qoff = (size_t)(g * 16 + kk * 4) * OUT_F;
            const uint32_t w00 = (uint32_t)qp0[qoff];
            const uint32_t w01 = (uint32_t)qp0[qoff + OUT_F];
            const uint32_t w10 = (uint32_t)qp1[qoff];
            const uint32_t w11 = (uint32_t)qp1[qoff + OUT_F];

            const v16h b0 = dequant16(w00, w01, s20);
            const v16h b1 = dequant16(w10, w11, s21);

            // ---- WMMA accumulate: 8 per K-step per wave ----
#pragma unroll
            for (int t = 0; t < 4; ++t) {
                acc[t][0] = __builtin_amdgcn_wmma_f32_16x16x32_f16(
                    false, a[t], false, b0, (short)0, acc[t][0], false, false);
                acc[t][1] = __builtin_amdgcn_wmma_f32_16x16x32_f16(
                    false, a[t], false, b1, (short)0, acc[t][1], false, false);
            }
        }
    }

    // ---- Epilogue. C/D layout: element r -> M = half*8 + r, N = ln ----
#pragma unroll
    for (int t = 0; t < 4; ++t) {
#pragma unroll
        for (int nt = 0; nt < 2; ++nt) {
            const int n = n0 + nt * 16 + ln;
#pragma unroll
            for (int r = 0; r < 8; ++r) {
                const int m = m0 + t * 16 + half * 8 + r;
                out[(size_t)m * OUT_F + n] = acc[t][nt][r];
            }
        }
    }
}

// ---------------------------------------------------------------------------
// Launch. Inputs in setup_inputs() order:
//   d_in[0] hidden_states fp16 [8,32,8192]
//   d_in[1] qweight       int32 [1024,8192]
//   d_in[2] scales        fp16 [64,8192]
//   d_in[3] perm          int64 [8192]
// d_out: float [256*8192]; d_ws: >= 4 MB for permuted activations.
// ---------------------------------------------------------------------------
extern "C" void kernel_launch(void* const* d_in, const int* in_sizes, int n_in,
                              void* d_out, int out_size, void* d_ws, size_t ws_size,
                              hipStream_t stream)
{
    (void)in_sizes; (void)n_in; (void)out_size; (void)ws_size;

    const _Float16*  hidden  = (const _Float16*)d_in[0];
    const int*       qweight = (const int*)d_in[1];
    const _Float16*  scales  = (const _Float16*)d_in[2];
    const long long* perm    = (const long long*)d_in[3];
    float*           out     = (float*)d_out;
    _Float16*        xp      = (_Float16*)d_ws;   // [256][8192] fp16 = 4 MB

    // 1) activation permutation gather
    gptq_permute_kernel<<<dim3((M_TOT * IN_F) / 256), dim3(256), 0, stream>>>(
        hidden, perm, xp);

    // 2) fused dequant + WMMA GEMM: grid (N/128, M/128) = (64, 2)
    gptq_wmma_gemm_kernel<<<dim3(OUT_F / 128, M_TOT / 128), dim3(256), 0, stream>>>(
        xp, qweight, scales, out);
}
